// MemorySystem_82136954569349
// MI455X (gfx1250) — compile-verified, matrix operand order checked by reading, not run on hardware
//
#include <hip/hip_runtime.h>

#define BATCH   1024
#define SROWS   65536
#define DIM     512
#define NSPLIT  8
#define SCHUNK  (SROWS / NSPLIT)   // 8192 memory rows per split
#define BN      64                 // memory rows per flash iteration
#define LDP     520                // padded LDS stride (halfs) for transpose tile (16B aligned)
#define PSTRIDE 72                 // padded LDS stride (halfs) for P tile (16B aligned rows)

typedef __attribute__((ext_vector_type(16))) __bf16 v16bf;
typedef __attribute__((ext_vector_type(8)))  float  v8f;

union Frag {
  uint4 u[2];
  v16bf v;
};

#define WMMA_BF16(A, B, C) \
  __builtin_amdgcn_wmma_f32_16x16x32_bf16(false, (A), false, (B), (short)0, (C), false, false)

__device__ __forceinline__ unsigned short f2bf(float f) {
  unsigned int u = __float_as_uint(f);
  u += 0x7FFFu + ((u >> 16) & 1u);           // round-to-nearest-even
  return (unsigned short)(u >> 16);
}
__device__ __forceinline__ unsigned int pack2(float a, float b) {
  return (unsigned int)f2bf(a) | ((unsigned int)f2bf(b) << 16);
}

// ---------------------------------------------------------------------------
// Preprocess memory bank: f32 (S x D) -> bf16 row-major mb, bf16 transposed
// mbT (D x S), and inverse row norms. 32 rows per 256-thread block.
// ---------------------------------------------------------------------------
__global__ void prep_mem(const float* __restrict__ mem,
                         unsigned short* __restrict__ mb,
                         unsigned short* __restrict__ mbT,
                         float* __restrict__ minv) {
  __shared__ __align__(16) unsigned short tile[32 * LDP];
  const int t   = threadIdx.x;      // 256 threads
  const int r   = t >> 3;           // local row 0..31
  const int seg = t & 7;            // 64-col segment 0..7
  const long row = (long)blockIdx.x * 32 + r;

  const float*    src  = mem + row * DIM + seg * 64;
  unsigned short* dstg = mb  + row * DIM + seg * 64;
  unsigned short* dstl = tile + r * LDP + seg * 64;

  float ss = 0.f;
#pragma unroll
  for (int i = 0; i < 64; i += 8) {
    float4 a = *(const float4*)(src + i);
    float4 b = *(const float4*)(src + i + 4);
    ss += a.x*a.x + a.y*a.y + a.z*a.z + a.w*a.w;
    ss += b.x*b.x + b.y*b.y + b.z*b.z + b.w*b.w;
    uint4 h;
    h.x = pack2(a.x, a.y); h.y = pack2(a.z, a.w);
    h.z = pack2(b.x, b.y); h.w = pack2(b.z, b.w);
    *(uint4*)(dstg + i) = h;
    *(uint4*)(dstl + i) = h;
  }
  // reduce sum-of-squares over the 8 segment lanes of this row
  ss += __shfl_xor(ss, 1, 32);
  ss += __shfl_xor(ss, 2, 32);
  ss += __shfl_xor(ss, 4, 32);
  if (seg == 0) minv[row] = 1.0f / fmaxf(sqrtf(ss), 1e-12f);

  __syncthreads();
  // transpose tile -> mbT[d][s]
  for (int d = t; d < DIM; d += 256) {
    unsigned int pk[16];
#pragma unroll
    for (int j = 0; j < 16; j++) {
      unsigned int lo = tile[(2 * j)     * LDP + d];
      unsigned int hi = tile[(2 * j + 1) * LDP + d];
      pk[j] = lo | (hi << 16);
    }
    size_t ob = ((size_t)d << 16) + ((size_t)blockIdx.x << 5); // d*65536 + blk*32
    uint4* dst = (uint4*)(mbT + ob);
    dst[0] = make_uint4(pk[0],  pk[1],  pk[2],  pk[3]);
    dst[1] = make_uint4(pk[4],  pk[5],  pk[6],  pk[7]);
    dst[2] = make_uint4(pk[8],  pk[9],  pk[10], pk[11]);
    dst[3] = make_uint4(pk[12], pk[13], pk[14], pk[15]);
  }
}

// ---------------------------------------------------------------------------
// Preprocess queries: f32 (B x D) -> bf16 xb + per-row scale 8/||x||.
// ---------------------------------------------------------------------------
__global__ void prep_x(const float* __restrict__ x,
                       unsigned short* __restrict__ xb,
                       float* __restrict__ xscale) {
  const int t   = threadIdx.x;
  const int r   = t >> 3;
  const int seg = t & 7;
  const long row = (long)blockIdx.x * 32 + r;

  const float*    src  = x  + row * DIM + seg * 64;
  unsigned short* dstg = xb + row * DIM + seg * 64;

  float ss = 0.f;
#pragma unroll
  for (int i = 0; i < 64; i += 8) {
    float4 a = *(const float4*)(src + i);
    float4 b = *(const float4*)(src + i + 4);
    ss += a.x*a.x + a.y*a.y + a.z*a.z + a.w*a.w;
    ss += b.x*b.x + b.y*b.y + b.z*b.z + b.w*b.w;
    uint4 h;
    h.x = pack2(a.x, a.y); h.y = pack2(a.z, a.w);
    h.z = pack2(b.x, b.y); h.w = pack2(b.z, b.w);
    *(uint4*)(dstg + i) = h;
  }
  ss += __shfl_xor(ss, 1, 32);
  ss += __shfl_xor(ss, 2, 32);
  ss += __shfl_xor(ss, 4, 32);
  if (seg == 0) xscale[row] = 8.0f / fmaxf(sqrtf(ss), 1e-12f); // temperature folded in
}

// ---------------------------------------------------------------------------
// Flash attention over one (16-query block, S split). 4 waves / 128 threads.
// Wave w owns QK N-tile w (16 memory rows of the 64-row block) and the
// D-slice [w*128, w*128+128) of the output accumulator.
// ---------------------------------------------------------------------------
__launch_bounds__(128, 1)
__global__ void flash(const unsigned short* __restrict__ xb,
                      const unsigned short* __restrict__ mb,
                      const unsigned short* __restrict__ mbT,
                      const float* __restrict__ xscale,
                      const float* __restrict__ minv,
                      float* __restrict__ pO,
                      float* __restrict__ pml) {
  __shared__ float wmax[4][16];
  __shared__ float wsum[4][16];
  __shared__ __align__(16) unsigned short pl[16 * PSTRIDE];

  const int tid  = threadIdx.x;
  const int w    = tid >> 5;
  const int lane = tid & 31;
  const int n    = lane & 15;    // fragment row (A) / column (B,C)
  const int hi   = lane >> 4;    // lane half
  const int qb   = blockIdx.y;   // query block
  const int sp   = blockIdx.x;   // S split

  // Preload all 16 A-fragments of this query block (K = 512 = 16 x 32).
  Frag xf[16];
  {
    const unsigned short* xrow = xb + (size_t)(qb * 16 + n) * DIM;
#pragma unroll
    for (int kk = 0; kk < 16; kk++) {
      xf[kk].u[0] = *(const uint4*)(xrow + kk * 32 + hi * 8);
      xf[kk].u[1] = *(const uint4*)(xrow + kk * 32 + 16 + hi * 8);
    }
  }
  // Per-lane row scales: C-tile row for VGPR v is (v + 8*hi).
  float xs[8];
#pragma unroll
  for (int v = 0; v < 8; v++) xs[v] = xscale[qb * 16 + v + 8 * hi];

  const v8f vzero = {0.f, 0.f, 0.f, 0.f, 0.f, 0.f, 0.f, 0.f};
  v8f acc[8];
#pragma unroll
  for (int t2 = 0; t2 < 8; t2++) acc[t2] = vzero;

  float run_m[8], run_l[8];
#pragma unroll
  for (int v = 0; v < 8; v++) { run_m[v] = -1e30f; run_l[v] = 0.f; }

  const int s_begin = sp * SCHUNK;
  for (int s0 = s_begin; s0 < s_begin + SCHUNK; s0 += BN) {
    const int scol = s0 + w * 16; // this wave's 16 memory rows

    // ---- QK: 16 WMMAs over K=512 ----
    v8f c = vzero;
    const unsigned short* brow = mb + (size_t)(scol + n) * DIM;
    __builtin_prefetch(brow + (size_t)BN * DIM, 0, 0);
#pragma unroll
    for (int kk = 0; kk < 16; kk++) {
      Frag bf;
      bf.u[0] = *(const uint4*)(brow + kk * 32 + hi * 8);
      bf.u[1] = *(const uint4*)(brow + kk * 32 + 16 + hi * 8);
      c = WMMA_BF16(xf[kk].v, bf.v, c);
    }
    const float sm = minv[scol + n];
    float lg[8], tm[8];
#pragma unroll
    for (int v = 0; v < 8; v++) {
      lg[v] = c[v] * xs[v] * sm;
      float x = lg[v];                       // row max within each 16-lane half
      x = fmaxf(x, __shfl_xor(x, 1, 32));
      x = fmaxf(x, __shfl_xor(x, 2, 32));
      x = fmaxf(x, __shfl_xor(x, 4, 32));
      x = fmaxf(x, __shfl_xor(x, 8, 32));
      tm[v] = x;
    }
    if (lane == 0 || lane == 16) {
#pragma unroll
      for (int v = 0; v < 8; v++) wmax[w][v + 8 * hi] = tm[v];
    }
    __syncthreads();

    // ---- online max/rescale (all waves compute identical run_m/run_l) ----
    float nm[8], fac[8];
#pragma unroll
    for (int v = 0; v < 8; v++) {
      const int r = v + 8 * hi;
      float mmax = run_m[v];
      mmax = fmaxf(mmax, wmax[0][r]);
      mmax = fmaxf(mmax, wmax[1][r]);
      mmax = fmaxf(mmax, wmax[2][r]);
      mmax = fmaxf(mmax, wmax[3][r]);
      nm[v]  = mmax;
      fac[v] = __expf(run_m[v] - mmax);
      run_m[v] = mmax;
      run_l[v] *= fac[v];
    }
#pragma unroll
    for (int t2 = 0; t2 < 8; t2++)
#pragma unroll
      for (int v = 0; v < 8; v++) acc[t2][v] = acc[t2][v] * fac[v];

    // ---- P = exp(logit - max), row partial sums, stage P as bf16 ----
    float p[8], psum[8];
#pragma unroll
    for (int v = 0; v < 8; v++) {
      p[v] = __expf(lg[v] - nm[v]);
      float s = p[v];
      s += __shfl_xor(s, 1, 32);
      s += __shfl_xor(s, 2, 32);
      s += __shfl_xor(s, 4, 32);
      s += __shfl_xor(s, 8, 32);
      psum[v] = s;
    }
    if (lane == 0 || lane == 16) {
#pragma unroll
      for (int v = 0; v < 8; v++) wsum[w][v + 8 * hi] = psum[v];
    }
#pragma unroll
    for (int v = 0; v < 8; v++)
      pl[(v + 8 * hi) * PSTRIDE + w * 16 + n] = f2bf(p[v]);
    __syncthreads();

#pragma unroll
    for (int v = 0; v < 8; v++) {
      const int r = v + 8 * hi;
      run_l[v] += wsum[0][r] + wsum[1][r] + wsum[2][r] + wsum[3][r];
    }

    // ---- PV: P (16x64) x M-tile (64 x 128-slice) = 16 WMMAs ----
    Frag pa0, pa1;
    {
      const unsigned short* prow = pl + n * PSTRIDE;
      pa0.u[0] = *(const uint4*)(prow + hi * 8);
      pa0.u[1] = *(const uint4*)(prow + 16 + hi * 8);
      pa1.u[0] = *(const uint4*)(prow + 32 + hi * 8);
      pa1.u[1] = *(const uint4*)(prow + 48 + hi * 8);
    }
#pragma unroll
    for (int t2 = 0; t2 < 8; t2++) {
      const int d = w * 128 + t2 * 16 + n;
      const unsigned short* vcol = mbT + ((size_t)d << 16) + s0;
      __builtin_prefetch(vcol + BN, 0, 0);
      Frag b0, b1;
      b0.u[0] = *(const uint4*)(vcol + hi * 8);
      b0.u[1] = *(const uint4*)(vcol + 16 + hi * 8);
      b1.u[0] = *(const uint4*)(vcol + 32 + hi * 8);
      b1.u[1] = *(const uint4*)(vcol + 48 + hi * 8);
      acc[t2] = WMMA_BF16(pa0.v, b0.v, acc[t2]);
      acc[t2] = WMMA_BF16(pa1.v, b1.v, acc[t2]);
    }
    __syncthreads(); // LDS reused next iteration
  }

  // ---- write partial O (unnormalized) and (m, l) for this split ----
  const size_t obase = ((size_t)qb * NSPLIT + sp) * 16 * DIM;
#pragma unroll
  for (int t2 = 0; t2 < 8; t2++)
#pragma unroll
    for (int v = 0; v < 8; v++)
      pO[obase + (size_t)(v + 8 * hi) * DIM + w * 128 + t2 * 16 + n] = acc[t2][v];

  if (w == 0 && (lane == 0 || lane == 16)) {
    const size_t mlb = ((size_t)qb * NSPLIT + sp) * 32;
#pragma unroll
    for (int v = 0; v < 8; v++) {
      pml[mlb + (v + 8 * hi)]      = run_m[v];
      pml[mlb + 16 + (v + 8 * hi)] = run_l[v];
    }
  }
}

// ---------------------------------------------------------------------------
// Combine the NSPLIT partials per query row (flash-decode reduction).
// ---------------------------------------------------------------------------
__global__ void combine(const float* __restrict__ pO,
                        const float* __restrict__ pml,
                        float* __restrict__ out) {
  const int b  = blockIdx.x;   // query row
  const int qb = b >> 4, r = b & 15;
  const int t  = threadIdx.x;  // 128 threads

  float mk[NSPLIT], lk[NSPLIT];
  float M = -1e30f;
#pragma unroll
  for (int k = 0; k < NSPLIT; k++) {
    const size_t mlb = ((size_t)qb * NSPLIT + k) * 32;
    mk[k] = pml[mlb + r];
    lk[k] = pml[mlb + 16 + r];
    M = fmaxf(M, mk[k]);
  }
  float wgt[NSPLIT], L = 0.f;
#pragma unroll
  for (int k = 0; k < NSPLIT; k++) {
    wgt[k] = __expf(mk[k] - M);
    L += lk[k] * wgt[k];
  }
  const float invL = 1.0f / fmaxf(L, 1e-30f);

  for (int d = t; d < DIM; d += 128) {
    float o = 0.f;
#pragma unroll
    for (int k = 0; k < NSPLIT; k++) {
      const size_t base = (((size_t)qb * NSPLIT + k) * 16 + r) * DIM;
      o += pO[base + d] * wgt[k];
    }
    out[(size_t)b * DIM + d] = o * invL;
  }
}

// ---------------------------------------------------------------------------
extern "C" void kernel_launch(void* const* d_in, const int* in_sizes, int n_in,
                              void* d_out, int out_size, void* d_ws, size_t ws_size,
                              hipStream_t stream) {
  (void)in_sizes; (void)n_in; (void)out_size;
  const float* x   = (const float*)d_in[0];  // (1024, 512)
  const float* mem = (const float*)d_in[1];  // (65536, 512)
  float* out = (float*)d_out;                // (1024, 512)

  char* ws = (char*)d_ws;
  size_t off = 0;
  unsigned short* xb  = (unsigned short*)(ws + off); off += (size_t)BATCH * DIM * 2;     // 1 MB
  unsigned short* mb  = (unsigned short*)(ws + off); off += (size_t)SROWS * DIM * 2;     // 64 MB
  unsigned short* mbT = (unsigned short*)(ws + off); off += (size_t)DIM * SROWS * 2;     // 64 MB
  float* xs   = (float*)(ws + off);          off += (size_t)BATCH * 4;                   // 4 KB
  float* minv = (float*)(ws + off);          off += (size_t)SROWS * 4;                   // 256 KB
  float* pO   = (float*)(ws + off);          off += (size_t)(BATCH/16) * NSPLIT * 16 * DIM * 4; // 16 MB
  float* pml  = (float*)(ws + off);          off += (size_t)(BATCH/16) * NSPLIT * 32 * 4;
  if (ws_size < off) return; // workspace too small; nothing safe to do

  prep_x  <<<BATCH / 32, 256, 0, stream>>>(x, xb, xs);
  prep_mem<<<SROWS / 32, 256, 0, stream>>>(mem, mb, mbT, minv);

  dim3 grid(NSPLIT, BATCH / 16);
  flash<<<grid, 128, 0, stream>>>(xb, mb, mbT, xs, minv, pO, pml);

  combine<<<BATCH, 128, 0, stream>>>(pO, pml, out);
}
